// HMGN_91018946937301
// MI455X (gfx1250) — matrix operand order repaired and checked by reading, not compiled, not run on hardware
//
#include <hip/hip_runtime.h>
#include <math.h>

#define B_   4
#define C_   2048
#define HP_  14
#define P_   (HP_*HP_)
#define NC_  200
#define WD_  300
#define INT_ 1024
#define NE_  800
#define NL_  3
#define BP_  (B_*P_)
#define M_   (B_*NC_)
#define MP_  896          // padded GEMM rows: 7 blocks * (4 waves * 32 rows)
#define EPS_ 1e-5f

typedef _Float16 h16;
typedef _Float16 h16x16 __attribute__((ext_vector_type(16)));
typedef float    f32x8  __attribute__((ext_vector_type(8)));

union Frag { h16x16 h; uint4 q[2]; };

#define WMMA(A, Bf, Cc) __builtin_amdgcn_wmma_f32_16x16x32_f16(false, (A).h, false, (Bf).h, (short)0, (Cc), false, false)

// ---------------- WMMA GEMM: Y[MP,N] = X[MP,K] @ W[N,K]^T  (fp16 in, fp32 out) --------
// Block = 4 waves; block tile = 128 (M) x 64 (N); each wave computes 32x64.
// B tile staged in LDS (double-buffered, shared); A fragments ping-pong in registers;
// k-loop manually unrolled x2 so no register rotation. Grid = (MP/128, N/64).
__global__ __launch_bounds__(128) void gemm_h16(const h16* __restrict__ X,
                                                const h16* __restrict__ W,
                                                float* __restrict__ Y,
                                                int N, int K)
{
    constexpr int LST = 40;                 // halves per B column row (32 + 8 pad)
    __shared__ h16 lds[2][64 * LST];

    const int tid  = threadIdx.x;
    const int lane = tid & 31;
    const int wv   = tid >> 5;
    const int r    = lane & 15, kh = lane >> 4;
    const int m0   = (blockIdx.x * 8 + wv * 2) << 4;   // wave rows m0..m0+31
    const int n0   = blockIdx.y << 6;

    // cooperative B staging: 128 threads x 32B = 64 cols x 64B (one 64x32 k-tile)
    const int ccol = tid >> 1, chalf = tid & 1;
    const h16* wsrc = W + (size_t)(n0 + ccol) * K + chalf * 16;
    h16* cdst0 = &lds[0][ccol * LST + chalf * 16];
    h16* cdst1 = &lds[1][ccol * LST + chalf * 16];

    const h16* xp0 = X + (size_t)(m0 + r) * K + (size_t)kh * 8;
    const h16* xp1 = xp0 + (size_t)16 * K;

    f32x8 acc[8] = {};

    // prologue: stage k-step 0 into lds[0]; load A(0)
    {
        uint4 t0 = *(const uint4*)(wsrc);
        uint4 t1 = *(const uint4*)(wsrc + 8);
        *(uint4*)(cdst0) = t0; *(uint4*)(cdst0 + 8) = t1;
    }
    Frag ae0, ae1, ao0, ao1;
    ae0.q[0] = *(const uint4*)(xp0);  ae0.q[1] = *(const uint4*)(xp0 + 16);
    ae1.q[0] = *(const uint4*)(xp1);  ae1.q[1] = *(const uint4*)(xp1 + 16);
    __syncthreads();

    const int steps = K >> 5;               // K/32, even (K % 64 == 0)
    for (int s = 0; s < steps; s += 2) {
        // ---- even step s : consume lds[0] + ae*, prefetch step s+1 ------------------
        uint4 p0, p1;
        {
            const h16* wn = wsrc + (size_t)(s + 1) * 32;
            p0 = *(const uint4*)(wn); p1 = *(const uint4*)(wn + 8);
            const h16* xn0 = xp0 + (size_t)(s + 1) * 32;
            const h16* xn1 = xp1 + (size_t)(s + 1) * 32;
            ao0.q[0] = *(const uint4*)(xn0); ao0.q[1] = *(const uint4*)(xn0 + 16);
            ao1.q[0] = *(const uint4*)(xn1); ao1.q[1] = *(const uint4*)(xn1 + 16);
        }
        {
            Frag b[4];
#pragma unroll
            for (int t = 0; t < 4; ++t) {
                const h16* lb = &lds[0][(t * 16 + r) * LST + kh * 8];
                b[t].q[0] = *(const uint4*)(lb);
                b[t].q[1] = *(const uint4*)(lb + 16);
            }
#pragma unroll
            for (int t = 0; t < 4; ++t) {
                acc[t]     = WMMA(ae0, b[t], acc[t]);
                acc[t + 4] = WMMA(ae1, b[t], acc[t + 4]);
            }
        }
        *(uint4*)(cdst1) = p0; *(uint4*)(cdst1 + 8) = p1;
        __syncthreads();

        // ---- odd step s+1 : consume lds[1] + ao*, prefetch step s+2 -----------------
        const bool more = (s + 2) < steps;
        if (more) {
            const h16* wn = wsrc + (size_t)(s + 2) * 32;
            p0 = *(const uint4*)(wn); p1 = *(const uint4*)(wn + 8);
            const h16* xn0 = xp0 + (size_t)(s + 2) * 32;
            const h16* xn1 = xp1 + (size_t)(s + 2) * 32;
            ae0.q[0] = *(const uint4*)(xn0); ae0.q[1] = *(const uint4*)(xn0 + 16);
            ae1.q[0] = *(const uint4*)(xn1); ae1.q[1] = *(const uint4*)(xn1 + 16);
        }
        {
            Frag b[4];
#pragma unroll
            for (int t = 0; t < 4; ++t) {
                const h16* lb = &lds[1][(t * 16 + r) * LST + kh * 8];
                b[t].q[0] = *(const uint4*)(lb);
                b[t].q[1] = *(const uint4*)(lb + 16);
            }
#pragma unroll
            for (int t = 0; t < 4; ++t) {
                acc[t]     = WMMA(ao0, b[t], acc[t]);
                acc[t + 4] = WMMA(ao1, b[t], acc[t + 4]);
            }
        }
        if (more) { *(uint4*)(cdst0) = p0; *(uint4*)(cdst0 + 8) = p1; }
        __syncthreads();
    }

    // store D: 2 m sub-tiles x 4 n sub-tiles
#pragma unroll
    for (int mi = 0; mi < 2; ++mi) {
#pragma unroll
        for (int t = 0; t < 4; ++t) {
            float* y = Y + (size_t)(m0 + mi * 16 + kh * 8) * N + n0 + t * 16 + r;
#pragma unroll
            for (int j = 0; j < 8; ++j) y[(size_t)j * N] = acc[mi * 4 + t][j];
        }
    }
}

// ---------------- small helpers -------------------------------------------------------
__global__ void f2h_k(const float* __restrict__ s, h16* __restrict__ d, int n) {
    int i = blockIdx.x * blockDim.x + threadIdx.x;
    if (i < n) d[i] = (h16)s[i];
}

__global__ void zero_k(float* __restrict__ d, int n) {
    int i = blockIdx.x * blockDim.x + threadIdx.x;
    if (i < n) d[i] = 0.f;
}

__global__ void zero_h_k(h16* __restrict__ d, int n) {
    int i = blockIdx.x * blockDim.x + threadIdx.x;
    if (i < n) d[i] = (h16)0.f;
}

// img [B,C,14,14] -> fmap [B,P,C] (f32 + f16)
__global__ void prep_fmap_k(const float* __restrict__ img, float* __restrict__ fm,
                            h16* __restrict__ fmh) {
    int idx = blockIdx.x * blockDim.x + threadIdx.x;   // B*P*C
    if (idx >= BP_ * C_) return;
    int c = idx % C_; int t = idx / C_; int p = t % P_; int b = t / P_;
    float v = img[((size_t)(b * C_ + c)) * P_ + p];
    fm[idx] = v; fmh[idx] = (h16)v;
}

// f_wd[n,i] = word[n,:] . W2[i,:]
__global__ void fwd_k(const float* __restrict__ word, const float* __restrict__ W2,
                      float* __restrict__ fwd) {
    int idx = blockIdx.x * blockDim.x + threadIdx.x;   // NC*INT
    if (idx >= NC_ * INT_) return;
    int i = idx % INT_, n = idx / INT_;
    const float* wr = word + (size_t)n * WD_;
    const float* w2 = W2 + (size_t)i * WD_;
    float s = 0.f;
    for (int w = 0; w < WD_; ++w) s += wr[w] * w2[w];
    fwd[idx] = s;
}

// w_eff[i] = sum_o Wa[o]*W3[o,i];  ebias = b3.Wa + ba
__global__ void weff_k(const float* __restrict__ W3, const float* __restrict__ Wa,
                       const float* __restrict__ b3, const float* __restrict__ ba,
                       float* __restrict__ weff, float* __restrict__ ebias) {
    int i = blockIdx.x * blockDim.x + threadIdx.x;
    if (i < INT_) {
        float s = 0.f;
        for (int o = 0; o < INT_; ++o) s += Wa[o] * W3[(size_t)o * INT_ + i];
        weff[i] = s;
    }
    if (blockIdx.x == 0 && threadIdx.x == 0) {
        float s = 0.f;
        for (int o = 0; o < INT_; ++o) s += b3[o] * Wa[o];
        ebias[0] = s + ba[0];
    }
}

__global__ void edge_idx_k(const float* __restrict__ Es, const float* __restrict__ Ee,
                           int* __restrict__ esi, int* __restrict__ eei) {
    int e = blockIdx.x * blockDim.x + threadIdx.x;
    if (e >= NE_) return;
    int a = 0, b = 0;
    for (int n = 0; n < NC_; ++n) {
        if (Es[(size_t)e * NC_ + n] > 0.5f) a = n;
        if (Ee[(size_t)e * NC_ + n] > 0.5f) b = n;
    }
    esi[e] = a; eei[e] = b;
}

// coef[b,n,p] = sum_i tanh(f_wh[bp,i]*f_wd[n,i])*w_eff[i] + ebias  (one wave per (bp,n))
__global__ __launch_bounds__(256) void coef_k(const float* __restrict__ fwh,
                                              const float* __restrict__ fwd,
                                              const float* __restrict__ weff,
                                              const float* __restrict__ ebias,
                                              float* __restrict__ coef) {
    int wid  = (blockIdx.x * blockDim.x + threadIdx.x) >> 5;
    int lane = threadIdx.x & 31;
    if (wid >= BP_ * NC_) return;
    int n = wid % NC_, bp = wid / NC_;
    const float* fh = fwh + (size_t)bp * INT_;
    const float* fd = fwd + (size_t)n * INT_;
    float s = 0.f;
    for (int i = lane; i < INT_; i += 32) s += tanhf(fh[i] * fd[i]) * weff[i];
    for (int off = 16; off; off >>= 1) s += __shfl_xor(s, off, 32);
    if (lane == 0) {
        int b = bp / P_, p = bp % P_;
        coef[((size_t)(b * NC_ + n)) * P_ + p] = s + ebias[0];
    }
}

// softmax over the P axis, one block per (b,n)
__global__ void softmax_k(float* __restrict__ coef) {
    __shared__ float red[256];
    float* v = coef + (size_t)blockIdx.x * P_;
    int t = threadIdx.x;
    float xv = (t < P_) ? v[t] : -3.4e38f;
    red[t] = xv; __syncthreads();
    for (int s = 128; s; s >>= 1) { if (t < s) red[t] = fmaxf(red[t], red[t + s]); __syncthreads(); }
    float mx = red[0]; __syncthreads();
    float e = (t < P_) ? expf(xv - mx) : 0.f;
    red[t] = e; __syncthreads();
    for (int s = 128; s; s >>= 1) { if (t < s) red[t] += red[t + s]; __syncthreads(); }
    float inv = 1.f / red[0];
    if (t < P_) v[t] = e * inv;
}

// x[b,n,c] = sum_p coef[b,n,p]*fmap[b,p,c]  (+ f16 copy)
__global__ void attn_agg_k(const float* __restrict__ coef, const float* __restrict__ fm,
                           float* __restrict__ x, h16* __restrict__ xh) {
    int idx = blockIdx.x * blockDim.x + threadIdx.x;   // M*C
    if (idx >= M_ * C_) return;
    int c = idx % C_; int t = idx / C_; int n = t % NC_; int b = t / NC_;
    const float* cf = coef + (size_t)(b * NC_ + n) * P_;
    const float* f  = fm + (size_t)b * P_ * C_ + c;
    float s = 0.f;
    for (int p = 0; p < P_; ++p) s += cf[p] * f[(size_t)p * C_];
    x[idx] = s; xh[idx] = (h16)s;
}

// gate/message scatter: Y4 layout [MP][4C] = [Ui|Uj|Vi|Vj]
__global__ void edge_scatter_k(const float* __restrict__ Y4, const int* __restrict__ esi,
                               const int* __restrict__ eei, const float* __restrict__ bv,
                               float* __restrict__ acc) {
    int idx = blockIdx.x * blockDim.x + threadIdx.x;   // B*NE*C
    if (idx >= B_ * NE_ * C_) return;
    int c = idx % C_; int t = idx / C_; int e = t % NE_; int b = t / NE_;
    int ms = b * NC_ + esi[e], me = b * NC_ + eei[e];
    float vix = Y4[(size_t)me * (4 * C_) + 2 * C_ + c];
    float vjx = Y4[(size_t)ms * (4 * C_) + 3 * C_ + c];
    float gate = 1.f / (1.f + expf(-(vix + vjx + bv[c])));
    float msg  = Y4[(size_t)ms * (4 * C_) + C_ + c];
    atomicAdd(&acc[(size_t)me * C_ + c], gate * msg);
}

__global__ void finish_h_k(const float* __restrict__ Y4, const float* __restrict__ acc,
                           const float* __restrict__ bu, float* __restrict__ h) {
    int idx = blockIdx.x * blockDim.x + threadIdx.x;   // M*C
    if (idx >= M_ * C_) return;
    int c = idx % C_; int m = idx / C_;
    h[idx] = Y4[(size_t)m * (4 * C_) + c] + acc[idx] + bu[c];
}

__global__ void bn_stats_k(const float* __restrict__ h, float* __restrict__ mean,
                           float* __restrict__ rstd) {
    int c = blockIdx.x * blockDim.x + threadIdx.x;
    if (c >= C_) return;
    float s = 0.f, s2 = 0.f;
    for (int m = 0; m < M_; ++m) { float v = h[(size_t)m * C_ + c]; s += v; s2 += v * v; }
    float mu = s * (1.f / (float)M_);
    float var = s2 * (1.f / (float)M_) - mu * mu;
    mean[c] = mu; rstd[c] = rsqrtf(var + EPS_);
}

__global__ void norm_relu_k(const float* __restrict__ h, const float* __restrict__ mean,
                            const float* __restrict__ rstd, const float* __restrict__ g,
                            const float* __restrict__ bet, h16* __restrict__ xh) {
    int idx = blockIdx.x * blockDim.x + threadIdx.x;
    if (idx >= M_ * C_) return;
    int c = idx % C_;
    float v = (h[idx] - mean[c]) * rstd[c] * g[c] + bet[c];
    v = fmaxf(v, 0.f);
    xh[idx] = (h16)v;
}

__global__ void norm_res_relu_k(const float* __restrict__ h, const float* __restrict__ mean,
                                const float* __restrict__ rstd, const float* __restrict__ g,
                                const float* __restrict__ bet, const float* __restrict__ xr,
                                float* __restrict__ x, h16* __restrict__ xh) {
    int idx = blockIdx.x * blockDim.x + threadIdx.x;
    if (idx >= M_ * C_) return;
    int c = idx % C_;
    float v = (h[idx] - mean[c]) * rstd[c] * g[c] + bet[c] + xr[idx];
    v = fmaxf(v, 0.f);
    x[idx] = v; xh[idx] = (h16)v;
}

// head: one wave per (b,n); outputs split per reference tuple layout
__global__ __launch_bounds__(256) void head_k(const float* __restrict__ x,
                                              const float* __restrict__ fcw,
                                              const float* __restrict__ fcb,
                                              float* __restrict__ out) {
    int wid  = (blockIdx.x * blockDim.x + threadIdx.x) >> 5;
    int lane = threadIdx.x & 31;
    if (wid >= B_ * NC_) return;
    int n = wid % NC_, b = wid / NC_;
    const float* xv = x + (size_t)(b * NC_ + n) * C_;
    const float* w  = fcw + (size_t)n * C_;
    float s = 0.f;
    for (int c = lane; c < C_; c += 32) s += xv[c] * w[c];
    for (int off = 16; off; off >>= 1) s += __shfl_xor(s, off, 32);
    if (lane == 0) {
        float o = s + fcb[n];
        float sg = 1.f / (1.f + expf(-o));
        if (n < 13)       out[b * 13 + n] = sg;
        else if (n < 51)  out[52 + b * 38 + (n - 13)] = sg;
        else { out[204 + b * 149 + (n - 51)] = o; out[800 + b * 149 + (n - 51)] = sg; }
    }
}

// --------------------------------------------------------------------------------------
extern "C" void kernel_launch(void* const* d_in, const int* in_sizes, int n_in,
                              void* d_out, int out_size, void* d_ws, size_t ws_size,
                              hipStream_t stream)
{
    (void)in_sizes; (void)n_in; (void)out_size; (void)ws_size;
    const float* img   = (const float*)d_in[0];
    const float* word  = (const float*)d_in[1];
    const float* Es    = (const float*)d_in[2];
    const float* Ee    = (const float*)d_in[3];
    const float* W1    = (const float*)d_in[4];
    const float* W2    = (const float*)d_in[5];
    const float* W3    = (const float*)d_in[6];
    const float* b3    = (const float*)d_in[7];
    const float* Wa    = (const float*)d_in[8];
    const float* ba    = (const float*)d_in[9];
    const float* Wg[9];
    for (int j = 0; j < 9; ++j) Wg[j] = (const float*)d_in[10 + j]; // Ui1,Uj1,Vi1,Vj1,Ui2,Uj2,Vi2,Vj2,Rw
    const float* bu1   = (const float*)d_in[19];
    const float* bv1   = (const float*)d_in[20];
    const float* bu2   = (const float*)d_in[21];
    const float* bv2   = (const float*)d_in[22];
    const float* bn1_b = (const float*)d_in[23];
    const float* bn2_b = (const float*)d_in[24];
    const float* bn1_g = (const float*)d_in[25];
    const float* bn2_g = (const float*)d_in[26];
    const float* fc_w  = (const float*)d_in[27];
    const float* fc_b  = (const float*)d_in[28];
    float* out = (float*)d_out;

    char* p = (char*)d_ws;
    auto alloc = [&](size_t bytes) -> char* {
        char* r = p; p += (bytes + 255) & ~(size_t)255; return r;
    };
    h16*   wbuf  = (h16*)  alloc((size_t)9 * C_ * C_ * 2);
    h16*   w1h   = (h16*)  alloc((size_t)INT_ * C_ * 2);
    float* fmap  = (float*)alloc((size_t)BP_ * C_ * 4);
    h16*   fmaph = (h16*)  alloc((size_t)MP_ * C_ * 2);
    float* fwh   = (float*)alloc((size_t)MP_ * INT_ * 4);
    float* fwd   = (float*)alloc((size_t)NC_ * INT_ * 4);
    float* weff  = (float*)alloc((size_t)INT_ * 4);
    float* ebias = (float*)alloc(256);
    float* coef  = (float*)alloc((size_t)B_ * NC_ * P_ * 4);
    float* x     = (float*)alloc((size_t)M_ * C_ * 4);
    h16*   xha   = (h16*)  alloc((size_t)MP_ * C_ * 2);
    h16*   xhb   = (h16*)  alloc((size_t)MP_ * C_ * 2);
    float* Y4    = (float*)alloc((size_t)MP_ * 4 * C_ * 4);
    float* xr    = (float*)alloc((size_t)MP_ * C_ * 4);
    float* acc   = (float*)alloc((size_t)M_ * C_ * 4);
    float* hbuf  = (float*)alloc((size_t)M_ * C_ * 4);
    float* mean  = (float*)alloc((size_t)C_ * 4);
    float* rstd  = (float*)alloc((size_t)C_ * 4);
    int*   esi   = (int*)  alloc((size_t)NE_ * 4);
    int*   eei   = (int*)  alloc((size_t)NE_ * 4);

    const int TB = 256;
    const int nMC    = M_ * C_;              // 1,638,400
    const int gMC    = nMC / TB;             // 6400
    const int gBPC   = (BP_ * C_) / TB;      // 6272
    const int gEdge  = (B_ * NE_ * C_) / TB; // 25600
    const int gCC    = (C_ * C_) / TB;       // 16384

    // ---- zero GEMM row padding (rows >= real M) ---------------------------------------
    zero_h_k<<<((MP_ - BP_) * C_) / TB, TB, 0, stream>>>(fmaph + (size_t)BP_ * C_, (MP_ - BP_) * C_);
    zero_h_k<<<((MP_ - M_) * C_) / TB, TB, 0, stream>>>(xha + (size_t)M_ * C_, (MP_ - M_) * C_);
    zero_h_k<<<((MP_ - M_) * C_) / TB, TB, 0, stream>>>(xhb + (size_t)M_ * C_, (MP_ - M_) * C_);

    // ---- semantic attention -----------------------------------------------------------
    prep_fmap_k<<<gBPC, TB, 0, stream>>>(img, fmap, fmaph);
    f2h_k<<<(INT_ * C_) / TB, TB, 0, stream>>>(W1, w1h, INT_ * C_);
    // f_wh = fmap @ W1^T : MP=896, N=1024, K=2048
    gemm_h16<<<dim3(MP_ / 128, INT_ / 64), 128, 0, stream>>>(fmaph, w1h, fwh, INT_, C_);
    fwd_k<<<(NC_ * INT_) / TB, TB, 0, stream>>>(word, W2, fwd);
    weff_k<<<INT_ / TB, TB, 0, stream>>>(W3, Wa, b3, ba, weff, ebias);
    edge_idx_k<<<(NE_ + TB - 1) / TB, TB, 0, stream>>>(Es, Ee, esi, eei);
    coef_k<<<(BP_ * NC_ * 32) / TB, TB, 0, stream>>>(fwh, fwd, weff, ebias, coef);
    softmax_k<<<B_ * NC_, TB, 0, stream>>>(coef);
    attn_agg_k<<<gMC, TB, 0, stream>>>(coef, fmap, x, xha);

    // ---- 3 GatedGNN layers ------------------------------------------------------------
    for (int l = 0; l < NL_; ++l) {
        for (int j = 0; j < 9; ++j)
            f2h_k<<<gCC, TB, 0, stream>>>(Wg[j] + (size_t)l * C_ * C_, wbuf + (size_t)j * C_ * C_, C_ * C_);

        // half 1: Y4 = x @ [Ui1|Uj1|Vi1|Vj1]^T : MP=896, N=8192, K=2048
        gemm_h16<<<dim3(MP_ / 128, (4 * C_) / 64), 128, 0, stream>>>(xha, wbuf, Y4, 4 * C_, C_);
        zero_k<<<gMC, TB, 0, stream>>>(acc, nMC);
        edge_scatter_k<<<gEdge, TB, 0, stream>>>(Y4, esi, eei, bv1 + (size_t)l * C_, acc);
        finish_h_k<<<gMC, TB, 0, stream>>>(Y4, acc, bu1 + (size_t)l * C_, hbuf);
        bn_stats_k<<<C_ / TB, TB, 0, stream>>>(hbuf, mean, rstd);
        norm_relu_k<<<gMC, TB, 0, stream>>>(hbuf, mean, rstd,
                                            bn1_g + (size_t)l * C_, bn1_b + (size_t)l * C_, xhb);

        // residual: xr = x_in @ Rw^T : MP=896, N=2048
        gemm_h16<<<dim3(MP_ / 128, C_ / 64), 128, 0, stream>>>(xha, wbuf + (size_t)8 * C_ * C_, xr, C_, C_);

        // half 2
        gemm_h16<<<dim3(MP_ / 128, (4 * C_) / 64), 128, 0, stream>>>(xhb, wbuf + (size_t)4 * C_ * C_, Y4, 4 * C_, C_);
        zero_k<<<gMC, TB, 0, stream>>>(acc, nMC);
        edge_scatter_k<<<gEdge, TB, 0, stream>>>(Y4, esi, eei, bv2 + (size_t)l * C_, acc);
        finish_h_k<<<gMC, TB, 0, stream>>>(Y4, acc, bu2 + (size_t)l * C_, hbuf);
        bn_stats_k<<<C_ / TB, TB, 0, stream>>>(hbuf, mean, rstd);
        norm_res_relu_k<<<gMC, TB, 0, stream>>>(hbuf, mean, rstd,
                                                bn2_g + (size_t)l * C_, bn2_b + (size_t)l * C_,
                                                xr, x, xha);
    }

    // ---- head ------------------------------------------------------------------------
    head_k<<<(B_ * NC_ * 32) / TB, TB, 0, stream>>>(x, fc_w, fc_b, out); // 100 blocks
}
